// E73MatrixNonlinearCell_55456617726095
// MI455X (gfx1250) — matrix operand order repaired and checked by reading, not compiled, not run on hardware
//
#include <hip/hip_runtime.h>
#include <hip/hip_bf16.h>

// ---------------------------------------------------------------------------
// Types for CDNA5 WMMA
// ---------------------------------------------------------------------------
typedef __attribute__((ext_vector_type(16))) __bf16 v16bf;
typedef __attribute__((ext_vector_type(2)))  __bf16 v2bf;
typedef __attribute__((ext_vector_type(8)))  float  v8f;
typedef __attribute__((ext_vector_type(8)))  unsigned int v8u;

union Frag {
    v16bf bf;
    v8u   u;
    uint4 q[2];
};

static constexpr int    T_STEPS    = 2048;
static constexpr int    BATCH      = 32;
static constexpr int    DIM        = 1024;
static constexpr int    NSTATE     = 64;
static constexpr int    M_TOTAL    = T_STEPS * BATCH;                 // 65536 rows
static constexpr int    NCOMB      = 4 * NSTATE;                      // 256 combined outputs
static constexpr size_t MAT_STRIDE = (size_t)M_TOTAL * NSTATE;        // floats per projection array
static constexpr size_t WBF_FLOATS = (size_t)NCOMB * DIM / 2;         // ushort storage in float units

// ---------------------------------------------------------------------------
// Helpers
// ---------------------------------------------------------------------------
// Hardware packed f32x2 -> bf16x2 conversion (single VALU op) when available.
static __device__ __forceinline__ unsigned pack_bf16(float lo, float hi) {
    union { v2bf b; unsigned u; } c;
#if __has_builtin(__builtin_amdgcn_cvt_pk_bf16_f32)
    c.b = __builtin_amdgcn_cvt_pk_bf16_f32(lo, hi);
#else
    c.b.x = (__bf16)lo;
    c.b.y = (__bf16)hi;
#endif
    return c.u;
}

static __device__ __forceinline__ unsigned short bf_scalar(float f) {
    union { v2bf b; unsigned u; } c;
#if __has_builtin(__builtin_amdgcn_cvt_pk_bf16_f32)
    c.b = __builtin_amdgcn_cvt_pk_bf16_f32(f, 0.0f);
#else
    c.b.x = (__bf16)f;
    c.b.y = (__bf16)0.0f;
#endif
    return (unsigned short)(c.u & 0xFFFFu);
}

// Single-instruction reciprocal (v_rcp_f32) instead of IEEE division chain.
static __device__ __forceinline__ float fast_rcp(float x) {
#if __has_builtin(__builtin_amdgcn_rcpf)
    return __builtin_amdgcn_rcpf(x);
#else
    return 1.0f / x;
#endif
}

static __device__ __forceinline__ float fast_sigmoid(float x) {
    return fast_rcp(1.0f + __expf(-x));
}

#if __has_builtin(__builtin_amdgcn_tanhf)
static __device__ __forceinline__ float fast_tanh(float x) {
    return __builtin_amdgcn_tanhf(x);
}
#else
static __device__ __forceinline__ float fast_tanh(float x) {
    float t = __expf(2.0f * x);
    return (t - 1.0f) * fast_rcp(t + 1.0f);
}
#endif

// ---------------------------------------------------------------------------
// Kernel 1: convert the 4 weight matrices [64,1024] fp32 -> combined bf16
//           [256,1024] (row-major, K contiguous) in workspace.
// ---------------------------------------------------------------------------
__global__ __launch_bounds__(256) void prep_weights(
        const float* __restrict__ Wk, const float* __restrict__ Wv,
        const float* __restrict__ Wq, const float* __restrict__ Wz,
        unsigned short* __restrict__ Wbf) {
    int idx = blockIdx.x * 256 + threadIdx.x;          // 0 .. 256*1024-1
    int nc  = idx >> 10;                               // combined row
    int d   = idx & 1023;
    const float* src = (nc < 64) ? Wk : (nc < 128) ? Wv : (nc < 192) ? Wq : Wz;
    float f = src[(size_t)(nc & 63) * DIM + d];
    Wbf[idx] = bf_scalar(f);
}

// ---------------------------------------------------------------------------
// Kernel 2: projection GEMM  C[65536 x 256] = bf16(x)[65536 x 1024] * Wbf^T
//           8 waves/WG: 2 (M) x 4 (N); wave tile 32x64; WG tile 64x256.
//           Epilogue scatters into k/v/q/z arrays, fusing sigmoid(z + b_z).
// ---------------------------------------------------------------------------
__global__ __launch_bounds__(256) void proj_gemm(
        const float* __restrict__ x,
        const unsigned short* __restrict__ Wbf,
        const float* __restrict__ bz,
        float* __restrict__ kvqz) {
    const int lane   = threadIdx.x & 31;
    const int wave   = threadIdx.x >> 5;
    const int wave_m = wave >> 2;          // 0..1
    const int wave_n = wave & 3;           // 0..3
    const int mbase  = blockIdx.x * 64 + wave_m * 32;
    const int nbase  = wave_n * 64;
    const int l15    = lane & 15;
    const int hi     = lane >> 4;          // half-wave select

    v8f acc[2][4];
#pragma unroll
    for (int mi = 0; mi < 2; ++mi)
#pragma unroll
        for (int ni = 0; ni < 4; ++ni)
            acc[mi][ni] = {};

    for (int kb = 0; kb < DIM; kb += 32) {
        // ---- A fragments: 16x32 bf16, built from fp32 x on the fly.
        // lanes 0-15: K = kb+0..7 and kb+16..23 ; lanes 16-31: +8 offset.
        Frag a[2];
#pragma unroll
        for (int mi = 0; mi < 2; ++mi) {
            int row = mbase + mi * 16 + l15;
            const float* xr = x + (size_t)row * DIM + kb + hi * 8;
            float4 r0 = *(const float4*)(xr);
            float4 r1 = *(const float4*)(xr + 4);
            float4 r2 = *(const float4*)(xr + 16);
            float4 r3 = *(const float4*)(xr + 20);
            a[mi].u[0] = pack_bf16(r0.x, r0.y);
            a[mi].u[1] = pack_bf16(r0.z, r0.w);
            a[mi].u[2] = pack_bf16(r1.x, r1.y);
            a[mi].u[3] = pack_bf16(r1.z, r1.w);
            a[mi].u[4] = pack_bf16(r2.x, r2.y);
            a[mi].u[5] = pack_bf16(r2.z, r2.w);
            a[mi].u[6] = pack_bf16(r3.x, r3.y);
            a[mi].u[7] = pack_bf16(r3.z, r3.w);
        }
        // ---- B fragments: 32x16 bf16; lane = column, K = kb+0..7 and kb+16..23 ; lanes 16-31: +8 offset.
        Frag bfr[4];
#pragma unroll
        for (int ni = 0; ni < 4; ++ni) {
            int col = nbase + ni * 16 + l15;
            const unsigned short* wp = Wbf + (size_t)col * DIM + kb + hi * 8;
            bfr[ni].q[0] = *(const uint4*)(wp);
            bfr[ni].q[1] = *(const uint4*)(wp + 16);
        }
#pragma unroll
        for (int mi = 0; mi < 2; ++mi)
#pragma unroll
            for (int ni = 0; ni < 4; ++ni)
                acc[mi][ni] = __builtin_amdgcn_wmma_f32_16x16x32_bf16(
                    false, a[mi].bf, false, bfr[ni].bf,
                    (short)0, acc[mi][ni], false, false);
    }

    // ---- Epilogue: C 16x16 f32 layout: lane holds column (l15), rows
    //      hi*8 .. hi*8+7 in the 8 accumulator VGPRs.
#pragma unroll
    for (int mi = 0; mi < 2; ++mi) {
#pragma unroll
        for (int ni = 0; ni < 4; ++ni) {
            int col  = nbase + ni * 16 + l15;     // 0..255
            int mat  = col >> 6;                  // 0=k 1=v 2=q 3=z
            int n    = col & 63;
            float bias = (mat == 3) ? bz[n] : 0.0f;
            float* dst = kvqz + (size_t)mat * MAT_STRIDE + n;
            int row0 = mbase + mi * 16 + hi * 8;
#pragma unroll
            for (int r = 0; r < 8; ++r) {
                float v = acc[mi][ni][r];
                if (mat == 3) v = fast_sigmoid(v + bias);
                dst[(size_t)(row0 + r) * NSTATE] = v;
            }
        }
    }
}

// ---------------------------------------------------------------------------
// Kernel 3: sequential scan.  One workgroup per batch; thread owns
//           S[row = tid>>2][ (tid&3)*16 .. +15 ] in registers.
// ---------------------------------------------------------------------------
union V16 { float4 v4[4]; float f[16]; };

__global__ __launch_bounds__(256) void scan_kernel(
        const float* __restrict__ S0,
        const float* __restrict__ kvqz,
        float* __restrict__ out) {
    const int b   = blockIdx.x;
    const int t   = threadIdx.x;
    const int row = t >> 2;
    const int cq  = (t & 3) * 16;

    const float* kA = kvqz;
    const float* vA = kvqz + MAT_STRIDE;
    const float* qA = kvqz + 2 * MAT_STRIDE;
    const float* zA = kvqz + 3 * MAT_STRIDE;

    float S[16];
    {
        const float* s0p = S0 + (size_t)b * 4096 + row * 64 + cq;
#pragma unroll
        for (int j = 0; j < 16; ++j) S[j] = s0p[j];
    }

    const size_t step_stride = (size_t)BATCH * NSTATE;   // 2048 floats / step

    for (int step = 0; step < T_STEPS; ++step) {
        size_t base = ((size_t)step * BATCH + b) * NSTATE;

        // Prefetch 8 steps ahead (global_prefetch_b8 hint, no counters).
        if (step + 8 < T_STEPS) {
            size_t pb = base + 8 * step_stride;
            __builtin_prefetch(kA + pb + cq, 0, 1);
            __builtin_prefetch(zA + pb + cq, 0, 1);
            __builtin_prefetch(qA + pb + cq, 0, 1);
            __builtin_prefetch(vA + pb + row, 0, 1);
        }

        V16 kx, zx, qx;
#pragma unroll
        for (int i = 0; i < 4; ++i) {
            kx.v4[i] = *(const float4*)(kA + base + cq + 4 * i);
            zx.v4[i] = *(const float4*)(zA + base + cq + 4 * i);
            qx.v4[i] = *(const float4*)(qA + base + cq + 4 * i);
        }
        float vv = vA[base + row];

        // S = tanh(S * z_col + v * k_col)
        float p = 0.0f;
#pragma unroll
        for (int j = 0; j < 16; ++j) {
            S[j] = fast_tanh(fmaf(S[j], zx.f[j], vv * kx.f[j]));
            p    = fmaf(S[j], qx.f[j], p);
        }
        // reduce across the 4 lanes covering this row
        p += __shfl_xor(p, 1, 32);
        p += __shfl_xor(p, 2, 32);

        if ((t & 3) == 0) {
            float o = p * fast_sigmoid(p);   // silu(p)
            out[base + row] = p * o;         // p * silu(p)
        }
    }

    // Final state -> d_out after the [T,B,64] output block.
    float* sf = out + MAT_STRIDE + (size_t)b * 4096 + row * 64 + cq;
#pragma unroll
    for (int j = 0; j < 16; ++j) sf[j] = S[j];
}

// ---------------------------------------------------------------------------
// Launch
// ---------------------------------------------------------------------------
extern "C" void kernel_launch(void* const* d_in, const int* in_sizes, int n_in,
                              void* d_out, int out_size, void* d_ws, size_t ws_size,
                              hipStream_t stream) {
    const float* x  = (const float*)d_in[0];
    const float* S0 = (const float*)d_in[1];
    const float* Wk = (const float*)d_in[2];
    const float* Wv = (const float*)d_in[3];
    const float* Wq = (const float*)d_in[4];
    const float* Wz = (const float*)d_in[5];
    const float* bz = (const float*)d_in[6];
    float* out = (float*)d_out;

    unsigned short* Wbf  = (unsigned short*)d_ws;
    float*          kvqz = (float*)d_ws + WBF_FLOATS;

    // 1) weights -> bf16 (256 rows x 1024)
    prep_weights<<<(NCOMB * DIM) / 256, 256, 0, stream>>>(Wk, Wv, Wq, Wz, Wbf);
    // 2) fused 4-way projection GEMM with WMMA bf16, sigmoid(z)+bias fused
    proj_gemm<<<M_TOTAL / 64, 256, 0, stream>>>(x, Wbf, bz, kvqz);
    // 3) recurrent scan, state in registers, one WG per batch
    scan_kernel<<<BATCH, 256, 0, stream>>>(S0, kvqz, out);
}
